// KanTtsSAMBERT_46755013984586
// MI455X (gfx1250) — matrix operand order repaired and checked by reading, not compiled, not run on hardware
//
#include <hip/hip_runtime.h>
#include <math.h>

// ---------------- CDNA5 WMMA types ----------------
typedef __attribute__((ext_vector_type(16))) _Float16 v16h;
typedef __attribute__((ext_vector_type(8)))  float    v8f;

// Model constants (match reference)
#define BATCH   4
#define SEQ_T   1024
#define NTOK    (BATCH * SEQ_T)      // 4096 tokens
#define DMODEL  256
#define DMEM    512
#define DIN     80
#define DINPAD  96                   // DIN padded to multiple of 32
#define DINNER  1024
#define DOUT    240
#define NLAYER  6
#define NHEAD   8
#define DHEAD   32

// Epilogue bitmask
#define EP_BIAS  1
#define EP_RELU  2
#define EP_RES   4
#define EP_MZ    8
#define EP_SC    16

// ---------------- wave32 reductions ----------------
__device__ __forceinline__ float wave_sum(float v) {
  #pragma unroll
  for (int m = 16; m >= 1; m >>= 1) v += __shfl_xor(v, m, 32);
  return v;
}
__device__ __forceinline__ float wave_max(float v) {
  #pragma unroll
  for (int m = 16; m >= 1; m >>= 1) v = fmaxf(v, __shfl_xor(v, m, 32));
  return v;
}

// ---------------- Weight packing: f32 [K][Nsrc] -> f16 panels [KP][Npack][32] ----------------
// Bp[kp][n][kk] = B[kp*32+kk][n], zero-padded in K and N. Reads coalesced over n.
__global__ __launch_bounds__(256) void pack_b_kernel(
    const float* __restrict__ B, _Float16* __restrict__ Bp,
    int K, int Nsrc, int Npack, int total)
{
  int idx = blockIdx.x * 256 + threadIdx.x;     // linear over (kp, kk, n)
  if (idx >= total) return;
  int n  = idx % Npack;
  int t  = idx / Npack;
  int kk = t & 31;
  int kp = t >> 5;
  int k  = kp * 32 + kk;
  float v = (k < K && n < Nsrc) ? B[(size_t)k * Nsrc + n] : 0.f;
  Bp[(((size_t)kp * Npack + n) << 5) + kk] = (_Float16)v;
}

// ---------------- Zero-pad A: [M,Ksrc] -> [M,Kdst] ----------------
__global__ __launch_bounds__(256) void pad_a_kernel(
    const float* __restrict__ A, float* __restrict__ Ap, int Ksrc, int Kdst, int total)
{
  int idx = blockIdx.x * 256 + threadIdx.x;
  if (idx >= total) return;
  int row = idx / Kdst, k = idx - row * Kdst;
  Ap[idx] = (k < Ksrc) ? A[(size_t)row * Ksrc + k] : 0.f;
}

// ---------------- WMMA GEMM with packed-f16 B ----------------
// Each wave: 16 rows x 32 cols (2 accumulators). Block = 4 waves -> 16 x 128.
// Grid: (Npack/128, M/16). K bound-check-free: KP panels, zero padded.
// Fragment layouts used by this kernel (wave32):
//   A (16x32 f16): lane L -> m=L&15, hi=L>>4; e -> K=e+16*hi  (contiguous 16 floats of the row)
//   B (32x16 f16): lane L -> n=L&15; e -> K=e+16*hi  (contiguous in packed [.,n,kk])
//   C/D (16x16 f32): VGPR r -> row = r + 8*hi, col = L&15
template <int EPI>
__global__ __launch_bounds__(128) void gemm_packed_kernel(
    const float* __restrict__ A, int lda,
    const _Float16* __restrict__ Bp,
    const float* __restrict__ bias,
    const float* Res, const unsigned char* __restrict__ Mask,
    float* C, int KP, int Npack, int Nout, float scale)
{
  const int lane = threadIdx.x & 31;
  const int wave = threadIdx.x >> 5;
  const int l16  = lane & 15;
  const int hi   = lane >> 4;

  const int mbase = blockIdx.y << 4;
  const int mrow  = mbase + l16;
  const int ncol0 = (blockIdx.x << 7) + (wave << 5) + l16;  // first 16-col tile
  // second tile at ncol0 + 16

  const float* __restrict__ arow = A + (size_t)mrow * lda;

  v8f acc[2] = {v8f{}, v8f{}};
  for (int kp = 0; kp < KP; ++kp) {
    const float* ap = arow + (kp << 5) + (hi << 4);
    const float4 a0 = *(const float4*)(ap);
    const float4 a1 = *(const float4*)(ap + 4);
    const float4 a2 = *(const float4*)(ap + 8);
    const float4 a3 = *(const float4*)(ap + 12);
    v16h af;
    af[0]  = (_Float16)a0.x; af[1]  = (_Float16)a0.y; af[2]  = (_Float16)a0.z; af[3]  = (_Float16)a0.w;
    af[4]  = (_Float16)a1.x; af[5]  = (_Float16)a1.y; af[6]  = (_Float16)a1.z; af[7]  = (_Float16)a1.w;
    af[8]  = (_Float16)a2.x; af[9]  = (_Float16)a2.y; af[10] = (_Float16)a2.z; af[11] = (_Float16)a2.w;
    af[12] = (_Float16)a3.x; af[13] = (_Float16)a3.y; af[14] = (_Float16)a3.z; af[15] = (_Float16)a3.w;

    const _Float16* bp = Bp + (((size_t)kp * Npack + ncol0) << 5) + (hi << 4);
    const v16h bf0 = *(const v16h*)(bp);
    const v16h bf1 = *(const v16h*)(bp + (16 << 5));   // next 16-col tile

    acc[0] = __builtin_amdgcn_wmma_f32_16x16x32_f16(false, af, false, bf0, (short)0, acc[0], false, false);
    acc[1] = __builtin_amdgcn_wmma_f32_16x16x32_f16(false, af, false, bf1, (short)0, acc[1], false, false);
  }

  #pragma unroll
  for (int c = 0; c < 2; ++c) {
    const int n = ncol0 + (c << 4);
    if (n >= Nout) continue;
    const float bn = (EPI & EP_BIAS) ? bias[n] : 0.f;
    #pragma unroll
    for (int r = 0; r < 8; ++r) {
      const int m = mbase + r + (hi << 3);
      float val = acc[c][r] + bn;
      if (EPI & EP_RELU) val = fmaxf(val, 0.f);
      if (EPI & EP_RES)  val += Res[(size_t)m * Nout + n];
      if (EPI & EP_SC)   val *= scale;
      if (EPI & EP_MZ)   { if (Mask[m]) val = 0.f; }
      C[(size_t)m * Nout + n] = val;
    }
  }
}

// ---------------- LayerNorm: one wave per 256-wide row ----------------
__global__ __launch_bounds__(256) void ln_kernel(
    const float* __restrict__ X, const float* __restrict__ G,
    const float* __restrict__ Bv, float* __restrict__ Y, float eps)
{
  const int row  = blockIdx.x * 8 + (threadIdx.x >> 5);
  const int lane = threadIdx.x & 31;
  const float* xr = X + (size_t)row * DMODEL;
  float v[8], s = 0.f, s2 = 0.f;
  #pragma unroll
  for (int t = 0; t < 8; ++t) {
    float u = xr[t * 32 + lane];
    v[t] = u; s += u; s2 += u * u;
  }
  s = wave_sum(s); s2 = wave_sum(s2);
  const float mu  = s * (1.f / DMODEL);
  float var = s2 * (1.f / DMODEL) - mu * mu;
  var = fmaxf(var, 0.f);
  const float rs = rsqrtf(var + eps);
  #pragma unroll
  for (int t = 0; t < 8; ++t) {
    const int n = t * 32 + lane;
    Y[(size_t)row * DMODEL + n] = (v[t] - mu) * rs * G[n] + Bv[n];
  }
}

// ---------------- Banded PNCA attention ----------------
// One block per (b,i); one wave per head. Branch 0: self keys j in [max(i-xbw,0), i] (Kx/Vx).
// Branch 1: memory keys j in [i, min(i+hbw, T-1)] (Kh/Vh). Output = ox + oh, layout [B,T,H*Dh].
__global__ __launch_bounds__(256) void attn_kernel(
    const float* __restrict__ Q,
    const float* __restrict__ Kx, const float* __restrict__ Vx,
    const float* __restrict__ Kh, const float* __restrict__ Vh,
    const unsigned char* __restrict__ Mask,
    const int* __restrict__ xbw_p, const int* __restrict__ hbw_p,
    float* __restrict__ Out)
{
  const int bi   = blockIdx.x;       // b*T + i
  const int b    = bi / SEQ_T;
  const int i    = bi - b * SEQ_T;
  const int head = threadIdx.x >> 5;
  const int lane = threadIdx.x & 31;

  __shared__ float qs[NHEAD][DHEAD];
  const int rowoff = bi * DMODEL + head * DHEAD;
  qs[head][lane] = Q[rowoff + lane];
  __syncthreads();

  const float scale = 0.17677669529663687f;  // 1/sqrt(32)
  const int   bbase = b * SEQ_T;
  const unsigned char* mb = Mask + bbase;
  const float* q = qs[head];

  const int xbw = xbw_p[0];
  const int hbw = hbw_p[0];
  int js[2], je[2];
  js[0] = (i - xbw > 0) ? (i - xbw) : 0;  je[0] = i;
  js[1] = i;  je[1] = (i + hbw < SEQ_T - 1) ? (i + hbw) : (SEQ_T - 1);

  float acc = 0.f;
  #pragma unroll
  for (int br = 0; br < 2; ++br) {
    const float* Kp = (br == 0) ? Kx : Kh;
    const float* Vp = (br == 0) ? Vx : Vh;
    const int jstart = js[br];
    int Lb = je[br] - jstart + 1;
    if (Lb > 64) Lb = 64;                 // bands are <=51 / <=31 here

    float s0 = -1e30f, s1 = -1e30f;       // big-negative avoids inf-inf NaN
    #pragma unroll
    for (int c = 0; c < 2; ++c) {
      const int jj = lane + (c << 5);
      if (jj < Lb) {
        const int j = jstart + jj;
        if (!mb[j]) {
          const float4* kr = (const float4*)(Kp + (size_t)(bbase + j) * DMODEL + head * DHEAD);
          float d = 0.f;
          #pragma unroll
          for (int t = 0; t < 8; ++t) {
            const float4 kv = kr[t];
            d += q[t * 4 + 0] * kv.x + q[t * 4 + 1] * kv.y
               + q[t * 4 + 2] * kv.z + q[t * 4 + 3] * kv.w;
          }
          const float sv = d * scale;
          if (c == 0) s0 = sv; else s1 = sv;
        }
      }
    }
    float mx = wave_max(fmaxf(s0, s1));
    const float e0 = __expf(s0 - mx);     // underflows to 0 for masked/out-of-band
    const float e1 = __expf(s1 - mx);
    const float inv = 1.f / wave_sum(e0 + e1);
    for (int jj = 0; jj < Lb; ++jj) {
      const float ev = (jj < 32) ? __shfl(e0, jj, 32) : __shfl(e1, jj - 32, 32);
      const float* vr = Vp + (size_t)(bbase + jstart + jj) * DMODEL + head * DHEAD;
      acc += ev * inv * vr[lane];
    }
  }
  Out[rowoff + lane] = acc;
}

// ---------------- host-side helpers ----------------
static inline int cdiv(int a, int b) { return (a + b - 1) / b; }

static void run_pack(const float* B, _Float16* Bp, int K, int Nsrc, int Npack, hipStream_t st)
{
  const int KP = cdiv(K, 32);
  const int total = KP * 32 * Npack;
  pack_b_kernel<<<dim3(cdiv(total, 256)), dim3(256), 0, st>>>(B, Bp, K, Nsrc, Npack, total);
}

static void run_gemm(int epi, const float* A, int lda, const _Float16* Bp, const float* bias,
                     const float* Res, const unsigned char* Mask, float* C,
                     int M, int K, int Npack, int Nout, float scale, hipStream_t st)
{
  const int KP = cdiv(K, 32);
  dim3 grid(Npack / 128, M / 16), blk(128);
  switch (epi) {
    case 0:
      gemm_packed_kernel<0><<<grid, blk, 0, st>>>(A, lda, Bp, bias, Res, Mask, C, KP, Npack, Nout, scale); break;
    case EP_BIAS:
      gemm_packed_kernel<EP_BIAS><<<grid, blk, 0, st>>>(A, lda, Bp, bias, Res, Mask, C, KP, Npack, Nout, scale); break;
    case EP_BIAS | EP_RELU:
      gemm_packed_kernel<EP_BIAS | EP_RELU><<<grid, blk, 0, st>>>(A, lda, Bp, bias, Res, Mask, C, KP, Npack, Nout, scale); break;
    case EP_BIAS | EP_RES | EP_MZ:
      gemm_packed_kernel<EP_BIAS | EP_RES | EP_MZ><<<grid, blk, 0, st>>>(A, lda, Bp, bias, Res, Mask, C, KP, Npack, Nout, scale); break;
    default:
      gemm_packed_kernel<EP_BIAS | EP_RES | EP_MZ | EP_SC><<<grid, blk, 0, st>>>(A, lda, Bp, bias, Res, Mask, C, KP, Npack, Nout, scale); break;
  }
}

static void run_ln(const float* X, const float* G, const float* Bv, float* Y,
                   float eps, hipStream_t st)
{
  ln_kernel<<<dim3(NTOK / 8), dim3(256), 0, st>>>(X, G, Bv, Y, eps);
}

extern "C" void kernel_launch(void* const* d_in, const int* in_sizes, int n_in,
                              void* d_out, int out_size, void* d_ws, size_t ws_size,
                              hipStream_t stream)
{
  (void)in_sizes; (void)n_in; (void)out_size; (void)ws_size;

  const float* input   = (const float*)d_in[0];   // [B,T,80]
  const float* memory  = (const float*)d_in[1];   // [B,T,512]
  const int*   xbw     = (const int*)d_in[2];
  const int*   hbw     = (const int*)d_in[3];
  const unsigned char* mask = (const unsigned char*)d_in[4];  // [B,T] bool
  const float* pw0 = (const float*)d_in[5];   const float* pb0 = (const float*)d_in[6];
  const float* pw1 = (const float*)d_in[7];   const float* pb1 = (const float*)d_in[8];
  const float* pfw = (const float*)d_in[9];   const float* pfb = (const float*)d_in[10];
  const float* ipw = (const float*)d_in[11];  const float* ipb = (const float*)d_in[12];
  const float* ln1g = (const float*)d_in[13]; const float* ln1b = (const float*)d_in[14];
  const float* wq  = (const float*)d_in[15];  const float* bq  = (const float*)d_in[16];
  const float* wkx = (const float*)d_in[17];  const float* bkx = (const float*)d_in[18];
  const float* wvx = (const float*)d_in[19];  const float* bvx = (const float*)d_in[20];
  const float* wkh = (const float*)d_in[21];  const float* bkh = (const float*)d_in[22];
  const float* wvh = (const float*)d_in[23];  const float* bvh = (const float*)d_in[24];
  const float* wo  = (const float*)d_in[25];  const float* bo  = (const float*)d_in[26];
  const float* ln2g = (const float*)d_in[27]; const float* ln2b = (const float*)d_in[28];
  const float* w1  = (const float*)d_in[29];  const float* b1  = (const float*)d_in[30];
  const float* w2  = (const float*)d_in[31];  const float* b2  = (const float*)d_in[32];
  const float* lnfg = (const float*)d_in[33]; const float* lnfb = (const float*)d_in[34];
  const float* outw = (const float*)d_in[35]; const float* outb = (const float*)d_in[36];

  const int M = NTOK;
  float* ws = (float*)d_ws;
  float* x     = ws + (size_t) 0 * M * DMODEL;
  float* xn    = ws + (size_t) 1 * M * DMODEL;
  float* qb    = ws + (size_t) 2 * M * DMODEL;
  float* kxb   = ws + (size_t) 3 * M * DMODEL;
  float* vxb   = ws + (size_t) 4 * M * DMODEL;
  float* khb   = ws + (size_t) 5 * M * DMODEL;
  float* vhb   = ws + (size_t) 6 * M * DMODEL;
  float* ab    = ws + (size_t) 7 * M * DMODEL;
  float* ff    = ws + (size_t) 8 * M * DMODEL;   // [M, 1024]
  float* inpad = ws + (size_t)12 * M * DMODEL;   // [M, 96]

  // f16 packed-weight scratch (32B aligned: all preceding sizes are multiples of 32B)
  _Float16* hp = (_Float16*)(inpad + (size_t)M * DINPAD);
  _Float16* P_pw0 = hp; hp += (size_t)DINPAD * DMODEL;          // 96x256
  _Float16* P_pw1 = hp; hp += (size_t)DMODEL * DMODEL;
  _Float16* P_pfw = hp; hp += (size_t)DMODEL * DMODEL;
  _Float16* P_ipw = hp; hp += (size_t)(DMEM + DMODEL) * DMODEL; // 768x256 (top 512 | bottom 256)
  _Float16* P_q   = hp; hp += (size_t)DMODEL * DMODEL;          // per-layer region, reused
  _Float16* P_kx  = hp; hp += (size_t)DMODEL * DMODEL;
  _Float16* P_vx  = hp; hp += (size_t)DMODEL * DMODEL;
  _Float16* P_kh  = hp; hp += (size_t)DMEM * DMODEL;
  _Float16* P_vh  = hp; hp += (size_t)DMEM * DMODEL;
  _Float16* P_o   = hp; hp += (size_t)DMODEL * DMODEL;
  _Float16* P_1   = hp; hp += (size_t)DMODEL * DINNER;
  _Float16* P_2   = hp; hp += (size_t)DINNER * DMODEL;
  _Float16* P_out = hp; hp += (size_t)DMODEL * DMODEL;          // 256x256 (240 padded)

  // ---- prenet (reuse q/kx/vx buffers before the layer loop) ----
  {
    const int total = M * DINPAD;
    pad_a_kernel<<<dim3(cdiv(total, 256)), dim3(256), 0, stream>>>(input, inpad, DIN, DINPAD, total);
  }
  run_pack(pw0, P_pw0, DIN, DMODEL, DMODEL, stream);
  run_pack(pw1, P_pw1, DMODEL, DMODEL, DMODEL, stream);
  run_pack(pfw, P_pfw, DMODEL, DMODEL, DMODEL, stream);
  run_pack(ipw, P_ipw, DMEM + DMODEL, DMODEL, DMODEL, stream);
  run_pack(outw, P_out, DMODEL, DOUT, DMODEL, stream);

  run_gemm(EP_BIAS | EP_RELU, inpad, DINPAD, P_pw0, pb0, nullptr, nullptr, qb,
           M, DIN, DMODEL, DMODEL, 1.f, stream);
  run_gemm(EP_BIAS | EP_RELU, qb, DMODEL, P_pw1, pb1, nullptr, nullptr, kxb,
           M, DMODEL, DMODEL, DMODEL, 1.f, stream);
  run_gemm(EP_BIAS, kxb, DMODEL, P_pfw, pfb, nullptr, nullptr, vxb,
           M, DMODEL, DMODEL, DMODEL, 1.f, stream);

  // ---- in_proj: concat([memory, h]) @ W + b -> mask-zero -> *sqrt(256)=16 (split-K) ----
  run_gemm(0, memory, DMEM, P_ipw, nullptr, nullptr, nullptr, ab,
           M, DMEM, DMODEL, DMODEL, 1.f, stream);                       // ab = memory @ W[0:512]
  run_gemm(EP_BIAS | EP_RES | EP_MZ | EP_SC, vxb, DMODEL,
           P_ipw + (size_t)DMEM * DMODEL,                               // W[512:768] panels
           ipb, ab, mask, x, M, DMODEL, DMODEL, DMODEL, 16.0f, stream);

  for (int l = 0; l < NLAYER; ++l) {
    const size_t wsq = (size_t)l * DMODEL * DMODEL;
    const size_t wsh = (size_t)l * DMEM * DMODEL;
    const size_t wsf = (size_t)l * DMODEL * DINNER;
    run_pack(wq  + wsq, P_q,  DMODEL, DMODEL, DMODEL, stream);
    run_pack(wkx + wsq, P_kx, DMODEL, DMODEL, DMODEL, stream);
    run_pack(wvx + wsq, P_vx, DMODEL, DMODEL, DMODEL, stream);
    run_pack(wkh + wsh, P_kh, DMEM,   DMODEL, DMODEL, stream);
    run_pack(wvh + wsh, P_vh, DMEM,   DMODEL, DMODEL, stream);
    run_pack(wo  + wsq, P_o,  DMODEL, DMODEL, DMODEL, stream);
    run_pack(w1  + wsf, P_1,  DMODEL, DINNER, DINNER, stream);
    run_pack(w2  + wsf, P_2,  DINNER, DMODEL, DMODEL, stream);

    run_ln(x, ln1g + l * DMODEL, ln1b + l * DMODEL, xn, 1e-5f, stream);
    run_gemm(EP_BIAS, xn, DMODEL, P_q,  bq  + l * DMODEL, nullptr, nullptr, qb,  M, DMODEL, DMODEL, DMODEL, 1.f, stream);
    run_gemm(EP_BIAS, xn, DMODEL, P_kx, bkx + l * DMODEL, nullptr, nullptr, kxb, M, DMODEL, DMODEL, DMODEL, 1.f, stream);
    run_gemm(EP_BIAS, xn, DMODEL, P_vx, bvx + l * DMODEL, nullptr, nullptr, vxb, M, DMODEL, DMODEL, DMODEL, 1.f, stream);
    run_gemm(EP_BIAS, memory, DMEM, P_kh, bkh + l * DMODEL, nullptr, nullptr, khb, M, DMEM, DMODEL, DMODEL, 1.f, stream);
    run_gemm(EP_BIAS, memory, DMEM, P_vh, bvh + l * DMODEL, nullptr, nullptr, vhb, M, DMEM, DMODEL, DMODEL, 1.f, stream);
    attn_kernel<<<dim3(NTOK), dim3(256), 0, stream>>>(qb, kxb, vxb, khb, vhb, mask, xbw, hbw, ab);
    run_gemm(EP_BIAS | EP_RES | EP_MZ, ab, DMODEL, P_o, bo + l * DMODEL, x, mask, x,
             M, DMODEL, DMODEL, DMODEL, 1.f, stream);
    run_ln(x, ln2g + l * DMODEL, ln2b + l * DMODEL, xn, 1e-5f, stream);
    run_gemm(EP_BIAS | EP_RELU, xn, DMODEL, P_1, b1 + (size_t)l * DINNER, nullptr, nullptr, ff,
             M, DMODEL, DINNER, DINNER, 1.f, stream);
    run_gemm(EP_BIAS | EP_RES | EP_MZ, ff, DINNER, P_2, b2 + l * DMODEL, x, mask, x,
             M, DINNER, DMODEL, DMODEL, 1.f, stream);
  }

  run_ln(x, lnfg, lnfb, xn, 1e-6f, stream);
  run_gemm(EP_BIAS, xn, DMODEL, P_out, outb, nullptr, nullptr,
           (float*)d_out, M, DMODEL, DMODEL, DOUT, 1.f, stream);
}
